// HyperButterflyMLP_59322088292573
// MI455X (gfx1250) — matrix operand order repaired and checked by reading, not compiled, not run on hardware
//
#include <hip/hip_runtime.h>
#include <hip/hip_bf16.h>

typedef unsigned int u32;

// CDNA5 WMMA vector types (wave32)
typedef __attribute__((ext_vector_type(16))) __bf16 v16bf;
typedef __attribute__((ext_vector_type(8)))  float  v8f;

union Frag16 { v16bf v; u32 u[8]; };

#define BATCH   8192
#define IN_DIM  4096
#define HID     2048
#define OUT_N   1000
#define LOG_DIM 11

// ---------------------------------------------------------------------------
// f32 -> split bf16 (hi = rne_bf16(x), lo = rne_bf16(x - hi)); packs 2 floats
// into one dword of hi parts and one dword of lo parts.
// ---------------------------------------------------------------------------
__device__ __forceinline__ void cvt2_split(float a, float b, u32& hid, u32& lod) {
    u32 ua = __float_as_uint(a);
    u32 ha = (ua + (0x7FFFu + ((ua >> 16) & 1u))) & 0xFFFF0000u;
    float la = a - __uint_as_float(ha);
    u32 ub = __float_as_uint(b);
    u32 hb = (ub + (0x7FFFu + ((ub >> 16) & 1u))) & 0xFFFF0000u;
    float lb = b - __uint_as_float(hb);
    hid = (ha >> 16) | hb;
    u32 ula = __float_as_uint(la); ula += 0x7FFFu + ((ula >> 16) & 1u);
    u32 ulb = __float_as_uint(lb); ulb += 0x7FFFu + ((ulb >> 16) & 1u);
    lod = (ula >> 16) | (ulb & 0xFFFF0000u);
}

// ---------------------------------------------------------------------------
// Split-bf16 WMMA GEMM:  C[M,N] = A[M,K] * B[N,K]^T + bias[N]
// Block tile 128x128, BK=32, 256 threads = 8 waves (4 x 2), each wave owns a
// 32x64 subtile (2x4 WMMA 16x16 accumulators). A*B ~= Ah*Bh + Ah*Bl + Al*Bh.
// LDS rows padded to 40 halfwords (20 dwords) -> conflict-free fragment reads.
// ---------------------------------------------------------------------------
#define LDSW 20   // dwords per 32-wide bf16 row (16 data + 4 pad)

template<bool NGUARD>
__global__ __launch_bounds__(256)
void gemm_split_bf16(const float* __restrict__ A, const float* __restrict__ B,
                     const float* __restrict__ bias, float* __restrict__ C,
                     int M, int N, int K)
{
    __shared__ __align__(16) u32 lAh[128 * LDSW];
    __shared__ __align__(16) u32 lAl[128 * LDSW];
    __shared__ __align__(16) u32 lBh[128 * LDSW];
    __shared__ __align__(16) u32 lBl[128 * LDSW];

    const int tid   = threadIdx.x;
    const int lane  = tid & 31;
    const int wave  = tid >> 5;
    const int wm    = wave & 3;       // 0..3 -> 32-row strip
    const int wn    = wave >> 2;      // 0..1 -> 64-col strip
    const int half  = lane >> 4;      // lane half (K-half selector)
    const int l16   = lane & 15;
    const int row0  = blockIdx.y * 128;
    const int col0  = blockIdx.x * 128;

    v8f acc[2][4] = {};

    for (int k0 = 0; k0 < K; k0 += 32) {
        // ---- stage A and B (f32 -> split bf16) into LDS -------------------
        #pragma unroll
        for (int pass = 0; pass < 2; ++pass) {
            int c  = tid + pass * 256;    // 0..511 chunk id (8 floats each)
            int r  = c >> 2;              // tile row 0..127
            int cg = c & 3;               // 8-float column group
            // A tile
            {
                const float* src = A + (size_t)(row0 + r) * K + k0 + cg * 8;
                if (k0 + 32 < K) __builtin_prefetch(src + 32, 0, 0);
                float4 f0 = ((const float4*)src)[0];
                float4 f1 = ((const float4*)src)[1];
                uint4 h, l;
                cvt2_split(f0.x, f0.y, h.x, l.x);
                cvt2_split(f0.z, f0.w, h.y, l.y);
                cvt2_split(f1.x, f1.y, h.z, l.z);
                cvt2_split(f1.z, f1.w, h.w, l.w);
                *((uint4*)&lAh[r * LDSW + cg * 4]) = h;
                *((uint4*)&lAl[r * LDSW + cg * 4]) = l;
            }
            // B tile (row r of the tile = output column col0+r)
            {
                int n = col0 + r;
                uint4 h = make_uint4(0, 0, 0, 0), l = make_uint4(0, 0, 0, 0);
                if (!NGUARD || n < N) {
                    const float* src = B + (size_t)n * K + k0 + cg * 8;
                    if (k0 + 32 < K) __builtin_prefetch(src + 32, 0, 0);
                    float4 f0 = ((const float4*)src)[0];
                    float4 f1 = ((const float4*)src)[1];
                    cvt2_split(f0.x, f0.y, h.x, l.x);
                    cvt2_split(f0.z, f0.w, h.y, l.y);
                    cvt2_split(f1.x, f1.y, h.z, l.z);
                    cvt2_split(f1.z, f1.w, h.w, l.w);
                }
                *((uint4*)&lBh[r * LDSW + cg * 4]) = h;
                *((uint4*)&lBl[r * LDSW + cg * 4]) = l;
            }
        }
        __syncthreads();

        // ---- fragment loads ----------------------------------------------
        // A 16x32 bf16 layout: lane half h, VGPR v<4: K=8h+2v ; v>=4: K=16+8h+2(v-4)
        Frag16 ah[2], al[2];
        #pragma unroll
        for (int i = 0; i < 2; ++i) {
            int r = 32 * wm + 16 * i + l16;
            const uint4* ph = (const uint4*)&lAh[r * LDSW];
            const uint4* pl = (const uint4*)&lAl[r * LDSW];
            uint4 h0 = ph[half], h1 = ph[2 + half];
            uint4 l0 = pl[half], l1 = pl[2 + half];
            ah[i].u[0]=h0.x; ah[i].u[1]=h0.y; ah[i].u[2]=h0.z; ah[i].u[3]=h0.w;
            ah[i].u[4]=h1.x; ah[i].u[5]=h1.y; ah[i].u[6]=h1.z; ah[i].u[7]=h1.w;
            al[i].u[0]=l0.x; al[i].u[1]=l0.y; al[i].u[2]=l0.z; al[i].u[3]=l0.w;
            al[i].u[4]=l1.x; al[i].u[5]=l1.y; al[i].u[6]=l1.z; al[i].u[7]=l1.w;
        }
        // B 32x16 bf16 layout (same as A): lane half h, VGPR v<4: K=8h+2v ;
        // v>=4: K=16+8h+2(v-4)
        Frag16 bh[4], bl[4];
        #pragma unroll
        for (int j = 0; j < 4; ++j) {
            int r = 64 * wn + 16 * j + l16;
            const uint4* ph = (const uint4*)&lBh[r * LDSW];
            const uint4* pl = (const uint4*)&lBl[r * LDSW];
            uint4 h0 = ph[half], h1 = ph[2 + half];
            uint4 l0 = pl[half], l1 = pl[2 + half];
            bh[j].u[0]=h0.x; bh[j].u[1]=h0.y; bh[j].u[2]=h0.z; bh[j].u[3]=h0.w;
            bh[j].u[4]=h1.x; bh[j].u[5]=h1.y; bh[j].u[6]=h1.z; bh[j].u[7]=h1.w;
            bl[j].u[0]=l0.x; bl[j].u[1]=l0.y; bl[j].u[2]=l0.z; bl[j].u[3]=l0.w;
            bl[j].u[4]=l1.x; bl[j].u[5]=l1.y; bl[j].u[6]=l1.z; bl[j].u[7]=l1.w;
        }

        // ---- matrix core: hi*hi + hi*lo + lo*hi ---------------------------
        #pragma unroll
        for (int i = 0; i < 2; ++i)
            #pragma unroll
            for (int j = 0; j < 4; ++j) {
                acc[i][j] = __builtin_amdgcn_wmma_f32_16x16x32_bf16(
                    false, ah[i].v, false, bh[j].v, (short)0, acc[i][j], false, false);
                acc[i][j] = __builtin_amdgcn_wmma_f32_16x16x32_bf16(
                    false, ah[i].v, false, bl[j].v, (short)0, acc[i][j], false, false);
                acc[i][j] = __builtin_amdgcn_wmma_f32_16x16x32_bf16(
                    false, al[i].v, false, bh[j].v, (short)0, acc[i][j], false, false);
            }
        __syncthreads();
    }

    // ---- epilogue: C/D layout -> lane l16 = N, VGPR r -> M = r + 8*half ---
    #pragma unroll
    for (int i = 0; i < 2; ++i) {
        int grow0 = row0 + 32 * wm + 16 * i + 8 * half;
        #pragma unroll
        for (int j = 0; j < 4; ++j) {
            int gcol = col0 + 64 * wn + 16 * j + l16;
            if (NGUARD && gcol >= N) continue;
            float bv = bias[gcol];
            #pragma unroll
            for (int r = 0; r < 8; ++r)
                C[(size_t)(grow0 + r) * N + gcol] = acc[i][j][r] + bv;
        }
    }
}

// ---------------------------------------------------------------------------
// BatchNorm stats: partial col sums over 256-row chunks, f32 global atomics
// ---------------------------------------------------------------------------
__global__ __launch_bounds__(256)
void zero_f32(float* p, int n) {
    int i = blockIdx.x * 256 + threadIdx.x;
    if (i < n) p[i] = 0.f;
}

__global__ __launch_bounds__(256)
void bn_stats(const float* __restrict__ h, float* __restrict__ sum,
              float* __restrict__ sumsq)
{
    int col = blockIdx.x * 256 + threadIdx.x;   // gridDim.x = 8
    int r0  = blockIdx.y * 256;                 // gridDim.y = 32
    float s = 0.f, s2 = 0.f;
    for (int r = 0; r < 256; ++r) {
        float v = h[(size_t)(r0 + r) * HID + col];
        s += v; s2 += v * v;
    }
    atomicAdd(&sum[col], s);
    atomicAdd(&sumsq[col], s2);
}

__global__ __launch_bounds__(256)
void bn_finalize(const float* __restrict__ sum, const float* __restrict__ sumsq,
                 float* __restrict__ mean, float* __restrict__ invstd)
{
    int c = blockIdx.x * 256 + threadIdx.x;
    if (c < HID) {
        float m = sum[c] * (1.f / BATCH);
        float v = sumsq[c] * (1.f / BATCH) - m * m;
        mean[c]   = m;
        invstd[c] = rsqrtf(v + 1e-5f);
    }
}

// ---------------------------------------------------------------------------
// Fused per-row: BN + ReLU + euclidean->Poincare + log map (origin) +
// butterfly (in LDS) + exp map (origin) + Poincare->euclidean
// ---------------------------------------------------------------------------
struct Bfly { const float* a[LOG_DIM]; const float* b[LOG_DIM]; };

__device__ __forceinline__ float blockReduceSum(float v, float* red, int tid) {
    red[tid] = v;
    __syncthreads();
    for (int off = 128; off > 0; off >>= 1) {
        if (tid < off) red[tid] += red[tid + off];
        __syncthreads();
    }
    float r = red[0];
    __syncthreads();
    return r;
}

__global__ __launch_bounds__(256)
void hyper_butterfly(const float* __restrict__ h, const float* __restrict__ mean,
                     const float* __restrict__ invstd, const float* __restrict__ gamma,
                     const float* __restrict__ beta, Bfly bf,
                     float* __restrict__ heuc)
{
    __shared__ float s[HID];
    __shared__ float red[256];
    const int tid = threadIdx.x;
    const int row = blockIdx.x;
    const float sqrt_c = 0.31622776601683794f;   // sqrt(0.1)
    const float EPS1   = 1.0f + 1e-8f;

    // BN + ReLU, accumulate ||.||^2
    float local = 0.f;
    for (int c = tid; c < HID; c += 256) {
        float v = h[(size_t)row * HID + c];
        v = (v - mean[c]) * invstd[c] * gamma[c] + beta[c];
        v = fmaxf(v, 0.f);
        s[c] = v;
        local += v * v;
    }
    float nsq  = blockReduceSum(local, red, tid);
    float norm = sqrtf(nsq);

    // euclidean_to_poincare: scale = 0.9*tanh(sc*n)/(sc*n)  (0 row -> 0)
    float safe   = (norm == 0.f) ? 1.f : norm;
    float pscale = 0.9f * tanhf(sqrt_c * norm) / (sqrt_c * safe);
    if (norm == 0.f) pscale = 0.f;
    float hn = pscale * norm;                       // ||h_hyp||
    // log map from origin (matches reference sign/EPS exactly):
    // v = -atanh(sc*dn)/(sc*dn) * h_hyp, dn = max(||h_hyp||/(1+eps), eps)
    float dn     = fmaxf(hn / EPS1, 1e-8f);
    float vscale = -atanhf(sqrt_c * dn) / (sqrt_c * dn);
    float total  = pscale * vscale;
    for (int c = tid; c < HID; c += 256) s[c] *= total;
    __syncthreads();

    // butterfly: 11 levels in LDS
    for (int l = 0; l < LOG_DIM; ++l) {
        const float* A = bf.a[l];
        const float* B = bf.b[l];
        int bs = 1 << l;
        for (int p = tid; p < HID / 2; p += 256) {
            int j  = p >> l;
            int t  = p & (bs - 1);
            int i0 = (j << (l + 1)) + t;
            int i1 = i0 + bs;
            float x0 = s[i0], x1 = s[i1];
            float av = A[j], bv = B[j];
            s[i0] =  av * x0 + bv * x1;
            s[i1] = -bv * x0 + av * x1;
        }
        __syncthreads();
    }

    // exp map from origin + poincare_to_euclidean
    local = 0.f;
    for (int c = tid; c < HID; c += 256) { float v = s[c]; local += v * v; }
    float vsq   = blockReduceSum(local, red, tid);
    float vnorm = sqrtf(vsq);
    float vn    = fmaxf(vnorm, 1e-8f);
    float lam   = 2.f / EPS1;
    float esc   = tanhf(sqrt_c * lam * vn * 0.5f) / (sqrt_c * vn);
    float et    = esc / EPS1;                       // h_t = et * v
    float htn   = et * vnorm;                       // ||h_t||
    float n2    = fminf(fmaxf(htn, 1e-8f), 1.f - 1e-8f);
    float fsc   = et * atanhf(sqrt_c * n2) / (sqrt_c * n2);
    for (int c = tid; c < HID; c += 256)
        heuc[(size_t)row * HID + c] = s[c] * fsc;
}

// ---------------------------------------------------------------------------
// In-place log-softmax over rows of 1000 (d_out fully rewritten by GEMM2
// each call, so in-place is graph-replay safe)
// ---------------------------------------------------------------------------
__global__ __launch_bounds__(256)
void logsoftmax_rows(float* __restrict__ out)
{
    __shared__ float red[256];
    const int row = blockIdx.x;
    const int tid = threadIdx.x;
    float* p = out + (size_t)row * OUT_N;

    float m = -3.402823466e+38f;
    for (int c = tid; c < OUT_N; c += 256) m = fmaxf(m, p[c]);
    red[tid] = m; __syncthreads();
    for (int off = 128; off > 0; off >>= 1) {
        if (tid < off) red[tid] = fmaxf(red[tid], red[tid + off]);
        __syncthreads();
    }
    m = red[0]; __syncthreads();

    float sum = 0.f;
    for (int c = tid; c < OUT_N; c += 256) sum += expf(p[c] - m);
    red[tid] = sum; __syncthreads();
    for (int off = 128; off > 0; off >>= 1) {
        if (tid < off) red[tid] += red[tid + off];
        __syncthreads();
    }
    sum = red[0]; __syncthreads();

    float lse = m + logf(sum);
    for (int c = tid; c < OUT_N; c += 256) p[c] = p[c] - lse;
}

// ---------------------------------------------------------------------------
extern "C" void kernel_launch(void* const* d_in, const int* in_sizes, int n_in,
                              void* d_out, int out_size, void* d_ws, size_t ws_size,
                              hipStream_t stream)
{
    const float* x     = (const float*)d_in[0];
    const float* w_in  = (const float*)d_in[1];
    const float* b_in  = (const float*)d_in[2];
    const float* gamma = (const float*)d_in[3];
    const float* beta  = (const float*)d_in[4];
    Bfly bf;
    const float* w_out;
    const float* b_out;
    if (n_in >= 5 + 2 * LOG_DIM + 2) {
        // tuples flattened: butterfly_a[0..10], butterfly_b[0..10]
        for (int l = 0; l < LOG_DIM; ++l) {
            bf.a[l] = (const float*)d_in[5 + l];
            bf.b[l] = (const float*)d_in[5 + LOG_DIM + l];
        }
        w_out = (const float*)d_in[5 + 2 * LOG_DIM];
        b_out = (const float*)d_in[6 + 2 * LOG_DIM];
    } else {
        // tuples concatenated into one buffer each (sizes 1024,512,...,1)
        const float* ba = (const float*)d_in[5];
        const float* bb = (const float*)d_in[6];
        for (int l = 0; l < LOG_DIM; ++l) {
            int off = 2048 - (2048 >> l);
            bf.a[l] = ba + off;
            bf.b[l] = bb + off;
        }
        w_out = (const float*)d_in[7];
        b_out = (const float*)d_in[8];
    }

    // workspace layout (bytes)
    char* ws = (char*)d_ws;
    float* h    = (float*)ws;                                  // 8192*2048 f32
    float* heuc = (float*)(ws + (size_t)BATCH * HID * 4);      // 8192*2048 f32
    float* ssum = (float*)(ws + (size_t)2 * BATCH * HID * 4);  // 2048
    float* ssq  = ssum + HID;                                  // 2048
    float* mean = ssq + HID;                                   // 2048
    float* invs = mean + HID;                                  // 2048

    // 1) GEMM1: h = x @ W_in^T + b_in   (split-bf16 WMMA)
    dim3 g1(HID / 128, BATCH / 128);
    gemm_split_bf16<false><<<g1, 256, 0, stream>>>(x, w_in, b_in, h,
                                                   BATCH, HID, IN_DIM);

    // 2) BatchNorm statistics
    zero_f32<<<(2 * HID + 255) / 256, 256, 0, stream>>>(ssum, 2 * HID);
    bn_stats<<<dim3(HID / 256, BATCH / 256), 256, 0, stream>>>(h, ssum, ssq);
    bn_finalize<<<(HID + 255) / 256, 256, 0, stream>>>(ssum, ssq, mean, invs);

    // 3) Fused BN/ReLU + hyperbolic maps + butterfly (per row)
    hyper_butterfly<<<BATCH, 256, 0, stream>>>(h, mean, invs, gamma, beta, bf, heuc);

    // 4) GEMM2: logits = heuc @ W_out^T + b_out  (into d_out, N guarded)
    dim3 g2((OUT_N + 127) / 128, BATCH / 128);
    gemm_split_bf16<true><<<g2, 256, 0, stream>>>(heuc, w_out, b_out,
                                                  (float*)d_out, BATCH, OUT_N, HID);

    // 5) In-place log-softmax
    logsoftmax_rows<<<BATCH, 256, 0, stream>>>((float*)d_out);
}
